// gat_lstm_autoencoder_42417097017254
// MI455X (gfx1250) — compile-verified, matrix operand order checked by reading, not run on hardware
//
#include <hip/hip_runtime.h>

// ---------------------------------------------------------------------------
// CDNA5 (gfx1250) implementation: GAT -> LSTM enc/dec -> GAT -> FC+MSE
// Big GEMMs (LSTM recurrence, FC) use v_wmma_f32_16x16x32_bf16, wave32.
// ---------------------------------------------------------------------------

typedef __attribute__((ext_vector_type(16))) __bf16 v16bf;
typedef __attribute__((ext_vector_type(8)))  float  v8f;

#define NUM_NODES 12
#define HEADS 4
#define PERIOD 24
#define HID 144
#define BTOT 98304
#define BB 4096          // BTOT / PERIOD
#define KPAD 160         // HID padded to multiple of 32
#define GDIM 576         // 4*HID gates

__device__ const int ADJd[12][12] = {
    {1,0,0,0,0,1,0,0,0,0,1,1},
    {0,1,0,0,0,0,0,1,0,0,1,0},
    {0,0,1,0,0,0,1,0,0,0,0,1},
    {0,0,0,1,0,0,1,0,0,0,0,0},
    {0,0,0,0,1,1,0,0,0,0,0,0},
    {1,0,0,0,1,1,0,0,0,0,0,0},
    {0,0,1,1,0,0,1,0,0,0,1,0},
    {0,1,0,0,0,0,0,1,0,1,0,0},
    {0,0,0,0,0,0,0,0,1,1,1,1},
    {0,0,0,0,0,0,0,1,1,1,1,0},
    {1,1,0,0,0,0,1,0,1,1,1,1},
    {1,0,1,0,0,0,0,0,1,0,1,1}};

// ---- helpers --------------------------------------------------------------

__device__ __forceinline__ unsigned short f2bf(float x) {
    unsigned int u = __float_as_uint(x);
    unsigned int r = u + 0x7fffu + ((u >> 16) & 1u);   // round-to-nearest-even
    return (unsigned short)(r >> 16);
}
__device__ __forceinline__ float sigm(float x) { return 1.0f / (1.0f + __expf(-x)); }

union FragU { uint4 u[2]; v16bf v; };

// Load one 16x32 bf16 WMMA operand fragment. Rows are K-major with leading
// dim `ld` (elements). Per ISA layout: lanes 0-15 -> K{k0..k0+7, k0+16..k0+23},
// lanes 16-31 -> K{k0+8..k0+15, k0+24..k0+31}; row index = row0 + (lane&15).
__device__ __forceinline__ v16bf load_frag(const unsigned short* __restrict__ base,
                                           int row0, int ld, int k0) {
    int lane = threadIdx.x & 31;
    const unsigned short* p =
        base + (size_t)(row0 + (lane & 15)) * ld + k0 + ((lane & 16) ? 8 : 0);
    FragU f;
    f.u[0] = *(const uint4*)(p);
    f.u[1] = *(const uint4*)(p + 16);
    return f.v;
}

__device__ __forceinline__ v8f wmma_bf16(v16bf a, v16bf b, v8f c) {
    return __builtin_amdgcn_wmma_f32_16x16x32_bf16(false, a, false, b,
                                                   (short)0, c, false, false);
}

// Store one 16x16 f32 C/D tile to LDS (ldg = row stride in floats).
// Lane l: col = ncol0+(l&15), rows m = v + ((l&16)?8:0) for v=0..7.
__device__ __forceinline__ void store_tile_lds(float* g, int ldg, int mrow0,
                                               int ncol0, v8f acc) {
    int lane = threadIdx.x & 31;
    int n = ncol0 + (lane & 15);
    int m0 = mrow0 + ((lane & 16) ? 8 : 0);
#pragma unroll
    for (int v = 0; v < 8; ++v) g[(m0 + v) * ldg + n] = acc[v];
}

// ---- weight conversion kernels -------------------------------------------

// src [N,Ks] f32 -> dst [N,Kd] bf16 (zero padded cols Ks..Kd)
__global__ void k_w_pad_bf16(const float* __restrict__ src,
                             unsigned short* __restrict__ dst,
                             int N, int Ks, int Kd) {
    int i = blockIdx.x * 256 + threadIdx.x;
    if (i >= N * Kd) return;
    int k = i % Kd, n = i / Kd;
    dst[i] = (k < Ks) ? f2bf(src[n * Ks + k]) : (unsigned short)0;
}

__global__ void k_add_bias(const float* __restrict__ a, const float* __restrict__ b,
                           float* __restrict__ o, int n) {
    int i = blockIdx.x * 256 + threadIdx.x;
    if (i < n) o[i] = a[i] + b[i];
}

// fW [288,264] f32 -> fcWt [272,288] bf16 transposed (rows 264..271 zero)
__global__ void k_fcw(const float* __restrict__ fw, unsigned short* __restrict__ dst) {
    int i = blockIdx.x * 256 + threadIdx.x;
    if (i >= 272 * 288) return;
    int k = i % 288, n = i / 288;
    dst[i] = (n < 264) ? f2bf(fw[k * 264 + n]) : (unsigned short)0;
}

// ---- GAT attention (sample 0 only) ---------------------------------------

template <int CIN, int HC, int H>
__global__ void k_attn(const float* __restrict__ in,   // sample0 rows [12,CIN]
                       const float* __restrict__ W,    // [CIN,HC]
                       const float* __restrict__ bias, // [HC]
                       const float* __restrict__ a,    // [H, 2*(HC/H)]
                       float* __restrict__ probs)      // [12*12*H]
{
    __shared__ float h0[12 * HC];
    for (int idx = threadIdx.x; idx < 12 * HC; idx += 64) {
        int hc = idx % HC, j = idx / HC;
        float s = bias[hc];
        for (int ci = 0; ci < CIN; ++ci) s += in[j * CIN + ci] * W[ci * HC + hc];
        h0[j * HC + hc] = s;
    }
    __syncthreads();
    const int CD = HC / H;
    int idx = threadIdx.x;
    if (idx < 12 * H) {
        int h = idx % H, i = idx / H;
        float lg[12];
        float mx = -3.0e38f;
        for (int j = 0; j < 12; ++j) {
            if (ADJd[i][j]) {
                float s = 0.f;
                for (int cc = 0; cc < CD; ++cc)
                    s += a[h * 2 * CD + cc]      * h0[i * HC + h * CD + cc]
                       + a[h * 2 * CD + CD + cc] * h0[j * HC + h * CD + cc];
                s = s > 0.f ? s : 0.2f * s;   // leaky 0.2
                lg[j] = s;
                if (s > mx) mx = s;
            } else lg[j] = -3.0e38f;
        }
        float den = 0.f, ex[12];
        for (int j = 0; j < 12; ++j) {
            float e = ADJd[i][j] ? __expf(lg[j] - mx) : 0.f;
            ex[j] = e; den += e;
        }
        float inv = 1.0f / den;
        for (int j = 0; j < 12; ++j) probs[(i * 12 + j) * H + h] = ex[j] * inv;
    }
}

// ---- fused GAT linear + aggregate + ELU ----------------------------------

template <int CIN, int HC, int H, bool OBF, int OLD>
__global__ __launch_bounds__(256) void k_gat(const float* __restrict__ in,
                                             const float* __restrict__ W,
                                             const float* __restrict__ bias,
                                             const float* __restrict__ probs,
                                             void* __restrict__ outv) {
    const int NB = 16;
    __shared__ float hl[NB * 12 * HC];
    __shared__ float pr[12 * 12 * H];
    __shared__ float Ws[CIN * HC + HC];
    int b0 = blockIdx.x * NB;
    for (int i = threadIdx.x; i < CIN * HC; i += 256) Ws[i] = W[i];
    for (int i = threadIdx.x; i < HC; i += 256) Ws[CIN * HC + i] = bias[i];
    for (int i = threadIdx.x; i < 144 * H; i += 256) pr[i] = probs[i];
    __syncthreads();
    for (int idx = threadIdx.x; idx < NB * 12 * HC; idx += 256) {
        int hc = idx % HC; int t = idx / HC; int j = t % 12; int bl = t / 12;
        const float* xr = in + ((size_t)(b0 + bl) * 12 + j) * CIN;
        float s = Ws[CIN * HC + hc];
        for (int ci = 0; ci < CIN; ++ci) s += xr[ci] * Ws[ci * HC + hc];
        hl[(bl * 12 + j) * HC + hc] = s;
    }
    __syncthreads();
    const int CD = HC / H;
    for (int idx = threadIdx.x; idx < NB * 12 * HC; idx += 256) {
        int hc = idx % HC; int t = idx / HC; int i = t % 12; int bl = t / 12;
        int h = hc / CD;
        float s = 0.f;
        for (int j = 0; j < 12; ++j)
            s += pr[(i * 12 + j) * H + h] * hl[(bl * 12 + j) * HC + hc];
        float e = s > 0.f ? s : (__expf(s) - 1.0f);   // ELU
        size_t row = (size_t)(b0 + bl);
        int col = i * HC + hc;
        if (OBF) ((unsigned short*)outv)[row * OLD + col] = f2bf(e);
        else     ((float*)outv)[row * OLD + col] = e;
    }
    if (OBF && OLD > 12 * HC) {   // zero K-pad columns for WMMA consumers
        const int P = OLD - 12 * HC;
        for (int idx = threadIdx.x; idx < NB * P; idx += 256) {
            int bl = idx / P, cc = idx % P;
            ((unsigned short*)outv)[(size_t)(b0 + bl) * OLD + 12 * HC + cc] = 0;
        }
    }
}

// ---- LSTM step: g = [xt|h] @ [Wx|Wh]^T + bias; gate math; in-place h,c ----
// block = 128 threads (4 waves), 16 batch rows; 36 WMMA tiles / 4 waves.

__global__ __launch_bounds__(128) void k_lstm_step(
    const unsigned short* __restrict__ xt,  // [BB,KPAD] bf16 or null
    const unsigned short* __restrict__ Wx,  // [GDIM,KPAD] bf16
    unsigned short* __restrict__ h,         // [BB,KPAD] bf16 (in/out, pads stay 0)
    const unsigned short* __restrict__ Wh,  // [GDIM,KPAD] bf16
    const float* __restrict__ bias,         // [GDIM] = bih + bhh
    float* __restrict__ c,                  // [BB,HID] f32 (in/out)
    unsigned short* __restrict__ ybf,       // [BB,KPAD] bf16 or null
    float* __restrict__ yf32)               // [BB,HID] f32 or null
{
    __shared__ float g[16 * GDIM];   // 36 KB
    int wave = threadIdx.x >> 5;
    int row0 = blockIdx.x * 16;
    int nt0 = wave * 9;

    v8f acc[9];
#pragma unroll
    for (int i = 0; i < 9; ++i)
#pragma unroll
        for (int e = 0; e < 8; ++e) acc[i][e] = 0.f;

    if (xt) {
#pragma unroll
        for (int kt = 0; kt < 5; ++kt) {
            v16bf a = load_frag(xt, row0, KPAD, kt * 32);
#pragma unroll
            for (int j = 0; j < 9; ++j) {
                v16bf b = load_frag(Wx, (nt0 + j) * 16, KPAD, kt * 32);
                acc[j] = wmma_bf16(a, b, acc[j]);
            }
        }
    }
#pragma unroll
    for (int kt = 0; kt < 5; ++kt) {
        v16bf a = load_frag(h, row0, KPAD, kt * 32);
#pragma unroll
        for (int j = 0; j < 9; ++j) {
            v16bf b = load_frag(Wh, (nt0 + j) * 16, KPAD, kt * 32);
            acc[j] = wmma_bf16(a, b, acc[j]);
        }
    }
#pragma unroll
    for (int j = 0; j < 9; ++j) store_tile_lds(g, GDIM, 0, (nt0 + j) * 16, acc[j]);
    __syncthreads();

    // torch gate order: i, f, g, o in column blocks of HID
    for (int idx = threadIdx.x; idx < 16 * HID; idx += 128) {
        int r = idx / HID, u = idx - r * HID;
        int R = row0 + r;
        float gi = g[r * GDIM + u]             + bias[u];
        float gf = g[r * GDIM + HID + u]       + bias[HID + u];
        float gg = g[r * GDIM + 2 * HID + u]   + bias[2 * HID + u];
        float go = g[r * GDIM + 3 * HID + u]   + bias[3 * HID + u];
        float cc = sigm(gf) * c[(size_t)R * HID + u] + sigm(gi) * tanhf(gg);
        float hh = sigm(go) * tanhf(cc);
        c[(size_t)R * HID + u] = cc;
        unsigned short hb = f2bf(hh);
        h[(size_t)R * KPAD + u] = hb;
        if (ybf)  ybf[(size_t)R * KPAD + u] = hb;
        if (yf32) yf32[(size_t)R * HID + u] = hh;
    }
}

// ---- FC (288->264) + leaky(0.01) + per-sample MSE vs x --------------------
// block = 256 threads (8 waves), 16 rows x 272 padded cols; 17 WMMA n-tiles.

__global__ __launch_bounds__(256) void k_fc_mse(
    const unsigned short* __restrict__ hin,  // [B,288] bf16
    const unsigned short* __restrict__ Wt,   // [272,288] bf16 (transposed fW)
    const float* __restrict__ fb,            // [264]
    const float* __restrict__ x,             // [B,264] f32
    float* __restrict__ out)                 // [B]
{
    __shared__ float g[16 * 272];
    __shared__ float red[256];
    int wave = threadIdx.x >> 5;
    int row0 = blockIdx.x * 16;

    for (int t = wave; t < 17; t += 8) {
        v8f acc;
#pragma unroll
        for (int e = 0; e < 8; ++e) acc[e] = 0.f;
#pragma unroll
        for (int kt = 0; kt < 9; ++kt) {
            v16bf a = load_frag(hin, row0, 288, kt * 32);
            v16bf b = load_frag(Wt, t * 16, 288, kt * 32);
            acc = wmma_bf16(a, b, acc);
        }
        int lane = threadIdx.x & 31;
        int n = t * 16 + (lane & 15);
        float bv = (n < 264) ? fb[n] : 0.f;
        int m0 = (lane & 16) ? 8 : 0;
#pragma unroll
        for (int v = 0; v < 8; ++v) {
            float val = acc[v] + bv;
            val = val > 0.f ? val : 0.01f * val;   // leaky 0.01
            g[(m0 + v) * 272 + n] = val;
        }
    }
    __syncthreads();

    int r = threadIdx.x >> 4, i0 = threadIdx.x & 15;
    const float* xr = x + (size_t)(row0 + r) * 264;
    float s = 0.f;
    for (int i = i0; i < 264; i += 16) {
        float d = xr[i] - g[r * 272 + i];
        s += d * d;
    }
    red[threadIdx.x] = s;
    __syncthreads();
    if (i0 == 0) {
        float tt = 0.f;
#pragma unroll
        for (int k = 0; k < 16; ++k) tt += red[(r << 4) + k];
        out[row0 + r] = tt * (1.0f / 264.0f);
    }
}

// ---------------------------------------------------------------------------

extern "C" void kernel_launch(void* const* d_in, const int* in_sizes, int n_in,
                              void* d_out, int out_size, void* d_ws, size_t ws_size,
                              hipStream_t stream) {
    const float* x     = (const float*)d_in[0];
    const float* gW0   = (const float*)d_in[1];
    const float* gb0   = (const float*)d_in[2];
    const float* ga0   = (const float*)d_in[3];
    const float* gW1   = (const float*)d_in[4];
    const float* gb1   = (const float*)d_in[5];
    const float* ga1   = (const float*)d_in[6];
    const float* lstmW[8] = {  // Wih/Whh in layer-pass order e0,e1,d0,d1
        (const float*)d_in[7],  (const float*)d_in[8],    // eWih0 eWhh0
        (const float*)d_in[11], (const float*)d_in[12],   // eWih1 eWhh1
        (const float*)d_in[15], (const float*)d_in[16],   // dWih0 dWhh0
        (const float*)d_in[19], (const float*)d_in[20]};  // dWih1 dWhh1
    const float* lstmB[8] = {
        (const float*)d_in[9],  (const float*)d_in[10],
        (const float*)d_in[13], (const float*)d_in[14],
        (const float*)d_in[17], (const float*)d_in[18],
        (const float*)d_in[21], (const float*)d_in[22]};
    const float* gW3 = (const float*)d_in[23];
    const float* gb3 = (const float*)d_in[24];
    const float* ga3 = (const float*)d_in[25];
    const float* gW4 = (const float*)d_in[26];
    const float* gb4 = (const float*)d_in[27];
    const float* ga4 = (const float*)d_in[28];
    const float* fW  = (const float*)d_in[29];
    const float* fb  = (const float*)d_in[30];

    char* ws = (char*)d_ws;
    size_t off = 0;
    auto alloc = [&](size_t sz) -> char* {
        off = (off + 255) & ~(size_t)255;
        char* p = ws + off;
        off += sz;
        return p;
    };

    unsigned short* Wbf[8];
    for (int i = 0; i < 8; ++i) Wbf[i] = (unsigned short*)alloc((size_t)GDIM * KPAD * 2);
    float* bsum[4];
    for (int i = 0; i < 4; ++i) bsum[i] = (float*)alloc(GDIM * 4);
    unsigned short* fcWt = (unsigned short*)alloc((size_t)272 * 288 * 2);
    float* probs[4];
    for (int i = 0; i < 4; ++i) probs[i] = (float*)alloc(12 * 12 * HEADS * 4);

    float* R1          = (float*)alloc((size_t)BTOT * 192 * 4);          // GAT0 out / y_final f32
    unsigned short* R2 = (unsigned short*)alloc((size_t)BTOT * KPAD * 2);// seq bf16 / dec-L0 y
    unsigned short* R5 = (unsigned short*)alloc((size_t)BTOT * KPAD * 2);// enc-L0 y bf16
    float* R3          = (float*)alloc((size_t)BTOT * 192 * 4);          // GAT3 out f32
    unsigned short* R4 = (unsigned short*)alloc((size_t)BTOT * 288 * 2); // GAT4 out bf16
    unsigned short* hA = (unsigned short*)alloc((size_t)BB * KPAD * 2);
    unsigned short* hB = (unsigned short*)alloc((size_t)BB * KPAD * 2);
    float* cA          = (float*)alloc((size_t)BB * HID * 4);
    float* cB          = (float*)alloc((size_t)BB * HID * 4);

    // --- weight conversion (deterministic, every call) ---
    for (int i = 0; i < 8; ++i)
        k_w_pad_bf16<<<(GDIM * KPAD + 255) / 256, 256, 0, stream>>>(lstmW[i], Wbf[i],
                                                                   GDIM, HID, KPAD);
    for (int l = 0; l < 4; ++l)
        k_add_bias<<<(GDIM + 255) / 256, 256, 0, stream>>>(lstmB[2 * l], lstmB[2 * l + 1],
                                                           bsum[l], GDIM);
    k_fcw<<<(272 * 288 + 255) / 256, 256, 0, stream>>>(fW, fcWt);

    // --- GAT0: [B,12,22] -> [B,12,16] f32 (R1) ---
    k_attn<22, 16, 4><<<1, 64, 0, stream>>>(x, gW0, gb0, ga0, probs[0]);
    k_gat<22, 16, 4, false, 192><<<BTOT / 16, 256, 0, stream>>>(x, gW0, gb0, probs[0], R1);

    // --- GAT1: [B,12,16] -> bf16 padded seq [B,160] (R2) ---
    k_attn<16, 12, 4><<<1, 64, 0, stream>>>(R1, gW1, gb1, ga1, probs[1]);
    k_gat<16, 12, 4, true, KPAD><<<BTOT / 16, 256, 0, stream>>>(R1, gW1, gb1, probs[1], R2);

    // --- LSTM state init ---
    hipMemsetAsync(hA, 0, (size_t)BB * KPAD * 2, stream);
    hipMemsetAsync(hB, 0, (size_t)BB * KPAD * 2, stream);
    hipMemsetAsync(cA, 0, (size_t)BB * HID * 4, stream);
    hipMemsetAsync(cB, 0, (size_t)BB * HID * 4, stream);
    hipMemsetAsync(R5, 0, (size_t)BTOT * KPAD * 2, stream);  // zero K-pads of enc-L0 y

    const size_t stepBF = (size_t)BB * KPAD;  // bf16 elems per timestep
    const size_t stepF  = (size_t)BB * HID;   // f32 elems per timestep
    dim3 lgrid(BB / 16), lblk(128);

    // encoder layer 0: seq(R2) -> y(R5)
    for (int t = 0; t < PERIOD; ++t)
        k_lstm_step<<<lgrid, lblk, 0, stream>>>(R2 + t * stepBF, Wbf[0], hA, Wbf[1],
                                                bsum[0], cA, R5 + t * stepBF, nullptr);
    // encoder layer 1: y(R5) -> (hn only)
    for (int t = 0; t < PERIOD; ++t)
        k_lstm_step<<<lgrid, lblk, 0, stream>>>(R5 + t * stepBF, Wbf[2], hB, Wbf[3],
                                                bsum[1], cB, nullptr, nullptr);
    // decoder: h carries over (hn), c reset to zero
    hipMemsetAsync(cA, 0, (size_t)BB * HID * 4, stream);
    hipMemsetAsync(cB, 0, (size_t)BB * HID * 4, stream);
    // decoder layer 0: zero input -> y(R2, pads already zero)
    for (int t = 0; t < PERIOD; ++t)
        k_lstm_step<<<lgrid, lblk, 0, stream>>>(nullptr, Wbf[4], hA, Wbf[5],
                                                bsum[2], cA, R2 + t * stepBF, nullptr);
    // decoder layer 1: y(R2) -> y_final f32 (R1 reuse)
    for (int t = 0; t < PERIOD; ++t)
        k_lstm_step<<<lgrid, lblk, 0, stream>>>(R2 + t * stepBF, Wbf[6], hB, Wbf[7],
                                                bsum[3], cB, nullptr, R1 + t * stepF);

    // --- GAT3: [B,12,12] f32 (R1) -> [B,12,16] f32 (R3) ---
    k_attn<12, 16, 4><<<1, 64, 0, stream>>>(R1, gW3, gb3, ga3, probs[2]);
    k_gat<12, 16, 4, false, 192><<<BTOT / 16, 256, 0, stream>>>(R1, gW3, gb3, probs[2], R3);

    // --- GAT4: [B,12,16] -> [B,288] bf16 (R4), K=288 = 9x32, no pad ---
    k_attn<16, 24, 4><<<1, 64, 0, stream>>>(R3, gW4, gb4, ga4, probs[3]);
    k_gat<16, 24, 4, true, 288><<<BTOT / 16, 256, 0, stream>>>(R3, gW4, gb4, probs[3], R4);

    // --- FC + leaky + per-sample MSE -> d_out [B] ---
    k_fc_mse<<<BTOT / 16, 256, 0, stream>>>(R4, fcWt, fb, x, (float*)d_out);

    (void)in_sizes; (void)n_in; (void)out_size; (void)ws_size;
}
